// SparseAttention_60043642798518
// MI455X (gfx1250) — matrix operand order repaired and checked by reading, not compiled, not run on hardware
//
#include <hip/hip_runtime.h>

#define H      8
#define DH     64
#define BLK    32
#define SEL    8
#define WIN    64
#define NMEM   4
#define DIM    512
#define NTOK   1024
#define NCB    32             /* NTOK/BLK */
#define NKC    (NMEM + NCB)   /* 36 */
#define NKCP   64             /* padded compressed-key count (WMMA K multiple) */
#define NB     (NTOK / WIN)   /* 16 */
#define SCALE  0.125f         /* DH^-0.5 */
#define NEGF   (-3.402823466e38f)

typedef __attribute__((ext_vector_type(16))) _Float16 v16h;
typedef __attribute__((ext_vector_type(8)))  float    v8f;
typedef _Float16 h16;

// ---------------------------------------------------------------------------
// WMMA helpers (v_wmma_f32_16x16x32_f16), A/B in contiguous k order (k = 16*(lane/16) + j)
// ---------------------------------------------------------------------------
__device__ __forceinline__ v8f wmma32(v16h a, v16h b, v8f c) {
  return __builtin_amdgcn_wmma_f32_16x16x32_f16(false, a, false, b, (short)0, c,
                                                false, false);
}

// A fragment: 16x32 f16, row-major source, leading dim lda (elements).
// lane<16: row=lane, K 0..15 ; lane>=16: row=lane-16, K 16..31
__device__ __forceinline__ v16h fragA_rm(const h16* base, int lda, int lane) {
  int r  = lane & 15;
  int k0 = (lane & 16) ? 16 : 0;
  const h16* p = base + r * lda + k0;
  v16h f;
#pragma unroll
  for (int j = 0; j < 16; ++j) f[j] = p[j];
  return f;
}

// B fragment from B^T stored row-major [N][K]: column n = lane&15, contiguous K.
__device__ __forceinline__ v16h fragB_cm(const h16* base, int ldb, int lane) {
  int n  = lane & 15;
  int k0 = (lane & 16) ? 16 : 0;
  const h16* p = base + n * ldb + k0;
  v16h f;
#pragma unroll
  for (int j = 0; j < 16; ++j) f[j] = p[j];
  return f;
}

// B fragment from B stored row-major [K][N] (strided gather per k).
__device__ __forceinline__ v16h fragB_rm(const h16* base, int ldb, int lane) {
  int n  = lane & 15;
  int k0 = (lane & 16) ? 16 : 0;
  v16h f;
#pragma unroll
  for (int j = 0; j < 16; ++j) f[j] = base[(k0 + j) * ldb + n];
  return f;
}

// C 16x16 f32 store: VGPR r -> row r + 8*(lane>=16), col = lane&15
__device__ __forceinline__ void storeC_f32(float* base, int ldc, v8f c, int lane) {
  int col = lane & 15;
  int r0  = (lane & 16) ? 8 : 0;
#pragma unroll
  for (int r = 0; r < 8; ++r) base[(r0 + r) * ldc + col] = c[r];
}

// ---------------------------------------------------------------------------
// CDNA5 async global->LDS copy (ASYNCcnt path). LDS operand is the low 32
// bits of the generic pointer (ISA 10.2).
// ---------------------------------------------------------------------------
__device__ __forceinline__ void async_g2l_b128(void* lds, const void* g) {
  asm volatile("global_load_async_to_lds_b128 %0, %1, off"
               :
               : "v"((unsigned)(unsigned long long)lds),
                 "v"((unsigned long long)g)
               : "memory");
}
__device__ __forceinline__ void wait_async0() {
  asm volatile("s_wait_asynccnt 0x0" ::: "memory");
}

// ---------------------------------------------------------------------------
// K0: convert weights to f16 (transposed so B-fragments load contiguously)
// ---------------------------------------------------------------------------
__global__ __launch_bounds__(256) void k_convert(
    const float* Wqkv, const float* Wout, const float* Wg,
    const float* kcw, const float* vcw,
    h16* Wqkvt, h16* Woutt, h16* Wgt, h16* kcwt, h16* vcwt) {
  int i = blockIdx.x * blockDim.x + threadIdx.x;
  int stride = gridDim.x * blockDim.x;
  for (int idx = i; idx < 3 * DIM * DIM; idx += stride) {   // Wqkvt[n][k]
    int n = idx / DIM, k = idx % DIM;
    Wqkvt[idx] = (h16)Wqkv[k * (3 * DIM) + n];
  }
  for (int idx = i; idx < DIM * DIM; idx += stride) {       // Woutt[n][k]
    int n = idx / DIM, k = idx % DIM;
    Woutt[idx] = (h16)Wout[k * DIM + n];
  }
  for (int idx = i; idx < 3 * H * DIM; idx += stride) {     // Wgt[n][k]
    int n = idx / DIM, k = idx % DIM;
    Wgt[idx] = (h16)Wg[k * (3 * H) + n];
  }
  // kcwt[h][o][t*DH+i2] = kc_w[h][o][i2][t]   (so conv becomes plain GEMM)
  for (int idx = i; idx < H * DH * DH * BLK; idx += stride) {
    int h   = idx / (DH * DH * BLK);
    int rem = idx % (DH * DH * BLK);
    int o   = rem / (DH * BLK);
    int k   = rem % (DH * BLK);
    int t = k / DH, i2 = k % DH;
    int src = ((h * DH + o) * DH + i2) * BLK + t;
    kcwt[idx] = (h16)kcw[src];
    vcwt[idx] = (h16)vcw[src];
  }
}

// ---------------------------------------------------------------------------
// K1: RMSNorm -> f16 xh
// ---------------------------------------------------------------------------
__global__ __launch_bounds__(128) void k_rmsnorm(const float* inp, const float* nw,
                                                 h16* xh) {
  int n = blockIdx.x, tid = threadIdx.x;
  __shared__ float red[4];
  const float* row = inp + n * DIM;
  float ss = 0.f;
  for (int c = tid; c < DIM; c += 128) { float v = row[c]; ss += v * v; }
  for (int off = 16; off; off >>= 1) ss += __shfl_down(ss, off, 32);
  if ((tid & 31) == 0) red[tid >> 5] = ss;
  __syncthreads();
  float rs = rsqrtf((red[0] + red[1] + red[2] + red[3]) / DIM + 1.1920929e-07f);
  for (int c = tid; c < DIM; c += 128)
    xh[n * DIM + c] = (h16)(row[c] * rs * nw[c]);
}

// ---------------------------------------------------------------------------
// K2: QKV GEMM (1024 x 1536 x 512) via WMMA; scatter into per-head q/k/v f16
// ---------------------------------------------------------------------------
__global__ __launch_bounds__(256) void k_gemm_qkv(const h16* xh, const h16* Wt,
                                                  h16* qh, h16* kh, h16* vh) {
  int lane = threadIdx.x & 31, wave = threadIdx.x >> 5;
  int m0 = blockIdx.x * 128 + (wave & 3) * 32;
  int n0 = blockIdx.y * 128 + (wave >> 2) * 64;
  v8f acc[2][4] = {};
  for (int k0 = 0; k0 < DIM; k0 += 32) {
    __builtin_prefetch(xh + m0 * DIM + k0 + 64, 0, 1);
    v16h a[2], b[4];
#pragma unroll
    for (int i2 = 0; i2 < 2; ++i2)
      a[i2] = fragA_rm(xh + (m0 + 16 * i2) * DIM + k0, DIM, lane);
#pragma unroll
    for (int j = 0; j < 4; ++j)
      b[j] = fragB_cm(Wt + (n0 + 16 * j) * DIM + k0, DIM, lane);
#pragma unroll
    for (int i2 = 0; i2 < 2; ++i2)
#pragma unroll
      for (int j = 0; j < 4; ++j) acc[i2][j] = wmma32(a[i2], b[j], acc[i2][j]);
  }
  int col0 = lane & 15, r0 = (lane & 16) ? 8 : 0;
#pragma unroll
  for (int i2 = 0; i2 < 2; ++i2)
#pragma unroll
    for (int j = 0; j < 4; ++j)
      for (int r = 0; r < 8; ++r) {
        int row = m0 + 16 * i2 + r0 + r;
        int col = n0 + 16 * j + col0;
        int part = col / DIM;
        int hh = (col % DIM) / DH;
        int d = col % DH;
        h16* dst = (part == 0) ? qh : ((part == 1) ? kh : vh);
        dst[(hh * NTOK + row) * DH + d] = (h16)acc[i2][j][r];
      }
}

// ---------------------------------------------------------------------------
// K3: mem-kv rows + zero padding of compressed key/value tables
// ---------------------------------------------------------------------------
__global__ __launch_bounds__(256) void k_fill_mem(const float* memkv, h16* ckh,
                                                  h16* cvh) {
  int h = blockIdx.x, tid = threadIdx.x;
  for (int e = tid; e < NMEM * DH; e += 256) {
    int j = e / DH, d = e % DH;
    ckh[(h * NKCP + j) * DH + d] = (h16)memkv[(0 * H + h) * NMEM * DH + e];
    cvh[(h * NKCP + j) * DH + d] = (h16)memkv[(1 * H + h) * NMEM * DH + e];
  }
  for (int e = tid; e < (NKCP - NKC) * DH; e += 256) {
    int j = NKC + e / DH, d = e % DH;
    ckh[(h * NKCP + j) * DH + d] = (h16)0.f;
    cvh[(h * NKCP + j) * DH + d] = (h16)0.f;
  }
}

// ---------------------------------------------------------------------------
// K4: compress conv as per-head GEMM  (32 x 64 x 2048), pos folded into A
//     waves 0-3 -> ck path, waves 4-7 -> cv path
// ---------------------------------------------------------------------------
__global__ __launch_bounds__(256) void k_compress(
    const h16* kh, const h16* vh, const float* kpos, const float* vpos,
    const float* kcb, const float* vcb, const h16* kcwt, const h16* vcwt,
    h16* ckh, h16* cvh) {
  int h = blockIdx.x;
  int tid = threadIdx.x, lane = tid & 31, wave = tid >> 5;
  int grp = wave >> 2, w4 = wave & 3;
  __shared__ h16 Atile[2][32][32];
  const h16* src = grp ? vh : kh;
  const float* pos = grp ? vpos : kpos;
  const h16* Bt = (grp ? vcwt : kcwt) + (size_t)h * DH * (DH * BLK);
  int mrow0 = (w4 & 1) * 16;
  int ncol0 = (w4 >> 1) * 32;
  v8f acc[2] = {};
  for (int k0 = 0; k0 < DH * BLK; k0 += 32) {
    int t = k0 / DH, i0 = k0 % DH;
    int gtid = w4 * 32 + lane;  // 0..127 within group
    for (int e = gtid; e < 32 * 32; e += 128) {
      int wrow = e >> 5, ic = e & 31;
      float val = (float)src[(h * NTOK + wrow * BLK + t) * DH + i0 + ic] +
                  pos[(h * BLK + t) * DH + i0 + ic];
      Atile[grp][wrow][ic] = (h16)val;
    }
    __syncthreads();
    v16h a = fragA_rm(&Atile[grp][mrow0][0], 32, lane);
#pragma unroll
    for (int j = 0; j < 2; ++j) {
      v16h b = fragB_cm(Bt + (ncol0 + 16 * j) * (DH * BLK) + k0, DH * BLK, lane);
      acc[j] = wmma32(a, b, acc[j]);
    }
    __syncthreads();
  }
  const float* bias = (grp ? vcb : kcb) + h * DH;
  h16* dst = (grp ? cvh : ckh) + h * NKCP * DH;
  int col0 = lane & 15, r0 = (lane & 16) ? 8 : 0;
#pragma unroll
  for (int j = 0; j < 2; ++j)
    for (int r = 0; r < 8; ++r) {
      int wrow = mrow0 + r0 + r;
      int o = ncol0 + 16 * j + col0;
      dst[(NMEM + wrow) * DH + o] = (h16)(acc[j][r] + bias[o]);
    }
}

// ---------------------------------------------------------------------------
// K5: csim = Q @ CK^T (scaled + masked), WMMA, M=1024 N=64 K=64 per head
// ---------------------------------------------------------------------------
__global__ __launch_bounds__(256) void k_csim(const h16* qh, const h16* ckh,
                                              float* csim) {
  int h = blockIdx.x, lane = threadIdx.x & 31, wave = threadIdx.x >> 5;
  int m0 = blockIdx.y * 128 + wave * 16;
  const h16* Q = qh + h * NTOK * DH;
  const h16* CK = ckh + h * NKCP * DH;
  v8f acc[4] = {};
  for (int k0 = 0; k0 < DH; k0 += 32) {
    v16h a = fragA_rm(Q + m0 * DH + k0, DH, lane);
#pragma unroll
    for (int j = 0; j < 4; ++j) {
      v16h b = fragB_cm(CK + (16 * j) * DH + k0, DH, lane);
      acc[j] = wmma32(a, b, acc[j]);
    }
  }
  int col0 = lane & 15, r0 = (lane & 16) ? 8 : 0;
  float* dst = csim + ((size_t)h * NTOK + m0) * NKCP;
#pragma unroll
  for (int j = 0; j < 4; ++j)
    for (int r = 0; r < 8; ++r) {
      int row = r0 + r;
      int col = 16 * j + col0;
      float v = acc[j][r] * SCALE;
      int tok = m0 + row;
      int ckseq = (col < NMEM) ? -1 : ((col - NMEM + 1) * BLK - 1);
      if (col >= NKC || ckseq < tok) v = NEGF;   // reference mask semantics
      dst[row * NKCP + col] = v;
    }
}

// ---------------------------------------------------------------------------
// K6: compressed softmax (f16 probs) + top-SEL block selection (one thr/row)
// ---------------------------------------------------------------------------
__global__ __launch_bounds__(256) void k_csoftmax_topk(const float* csim, h16* catt,
                                                       int* selidx,
                                                       unsigned* selflag) {
  int idx = blockIdx.x * blockDim.x + threadIdx.x;
  if (idx >= H * NTOK) return;
  const float* row = csim + (size_t)idx * NKCP;
  float m = NEGF;
  for (int j = 0; j < NKC; ++j) m = fmaxf(m, row[j]);
  float s = 0.f;
  for (int j = 0; j < NKC; ++j) s += __expf(row[j] - m);
  float inv = 1.f / s;
  h16* crow = catt + (size_t)idx * NKCP;
  for (int j = 0; j < NKC; ++j) crow[j] = (h16)(__expf(row[j] - m) * inv);
  for (int j = NKC; j < NKCP; ++j) crow[j] = (h16)0.f;
  unsigned chosen = 0, flags = 0;
  for (int s2 = 0; s2 < SEL; ++s2) {
    float best = NEGF * 2.f;  // -inf
    int bi = 0;
    for (int j = 0; j < NCB; ++j) {
      if (chosen & (1u << j)) continue;
      float v = row[NMEM + j];
      if (v > best) { best = v; bi = j; }
    }
    chosen |= 1u << bi;
    selidx[idx * SEL + s2] = bi;
    if (best > NEGF) flags |= 1u << s2;  // reference: fmask = sel_vals > NEG
  }
  selflag[idx] = flags;
}

// ---------------------------------------------------------------------------
// K7: compressed out = attn @ CV  (M=1024 N=64 K=64), WMMA
// ---------------------------------------------------------------------------
__global__ __launch_bounds__(256) void k_cout(const h16* catt, const h16* cvh,
                                              float* cout) {
  int h = blockIdx.x, lane = threadIdx.x & 31, wave = threadIdx.x >> 5;
  int m0 = blockIdx.y * 128 + wave * 16;
  const h16* A = catt + ((size_t)h * NTOK + m0) * NKCP;
  const h16* CV = cvh + h * NKCP * DH;  // B row-major [K][N]
  v8f acc[4] = {};
  for (int k0 = 0; k0 < NKCP; k0 += 32) {
    v16h a = fragA_rm(A + k0, NKCP, lane);
#pragma unroll
    for (int j = 0; j < 4; ++j) {
      v16h b = fragB_rm(CV + k0 * DH + 16 * j, DH, lane);
      acc[j] = wmma32(a, b, acc[j]);
    }
  }
  float* dst = cout + ((size_t)h * NTOK + m0) * DH;
#pragma unroll
  for (int j = 0; j < 4; ++j) storeC_f32(dst + 16 * j, DH, acc[j], lane);
}

// ---------------------------------------------------------------------------
// K8: fine attention — one wave per (head, token); gathered blocks, VALU
// ---------------------------------------------------------------------------
__global__ __launch_bounds__(256) void k_fine(const h16* qh, const h16* kh,
                                              const h16* vh, const int* selidx,
                                              const unsigned* selflag, float* fout) {
  int h = blockIdx.x;
  int wave = threadIdx.x >> 5, lane = threadIdx.x & 31;
  int n = blockIdx.y * 8 + wave;
  __shared__ float qs[8][DH];
  __shared__ float sims[8][SEL * BLK];
  __shared__ int keys[8][SEL];
  const h16* Q = qh + ((size_t)h * NTOK + n) * DH;
  qs[wave][2 * lane]     = (float)Q[2 * lane];
  qs[wave][2 * lane + 1] = (float)Q[2 * lane + 1];
  if (lane < SEL) keys[wave][lane] = selidx[((size_t)h * NTOK + n) * SEL + lane];
  unsigned flags = selflag[h * NTOK + n];
  __syncthreads();
  const h16* K = kh + (size_t)h * NTOK * DH;
  for (int s = 0; s < SEL; ++s) {
    int blkidx = keys[wave][s];
    const h16* krow = K + (blkidx * BLK + lane) * DH;
    float acc = 0.f;
#pragma unroll 8
    for (int d = 0; d < DH; ++d) acc += qs[wave][d] * (float)krow[d];
    acc *= SCALE;
    if (flags & (1u << s)) acc = NEGF;  // reference: mask where sel_val > NEG
    sims[wave][s * BLK + lane] = acc;
  }
  __syncthreads();
  float m = NEGF;
  for (int j = 0; j < SEL * BLK; ++j) m = fmaxf(m, sims[wave][j]);
  float ssum = 0.f;
  for (int j = 0; j < SEL * BLK; ++j) ssum += __expf(sims[wave][j] - m);
  float inv = 1.f / ssum;
  const h16* V = vh + (size_t)h * NTOK * DH;
  float o0 = 0.f, o1 = 0.f;
  for (int j = 0; j < SEL * BLK; ++j) {
    float p = __expf(sims[wave][j] - m) * inv;
    int key = keys[wave][j >> 5] * BLK + (j & 31);
    const h16* vrow = V + key * DH;
    o0 += p * (float)vrow[2 * lane];
    o1 += p * (float)vrow[2 * lane + 1];
  }
  float* dst = fout + ((size_t)h * NTOK + n) * DH;
  dst[2 * lane] = o0;
  dst[2 * lane + 1] = o1;
}

// ---------------------------------------------------------------------------
// K9: local window attention — async K/V staging to LDS (ASYNCcnt path),
//     WMMA sim, LDS softmax, WMMA attn@V from LDS.
//     LDS 64KB: kws/attw 16K | simw 32K | vws 16K
// ---------------------------------------------------------------------------
__global__ __launch_bounds__(256) void k_local(const h16* qh, const h16* kh,
                                               const h16* vh, float* lout) {
  __shared__ __align__(16) unsigned char smem_raw[65536];
  h16* kws    = (h16*)smem_raw;                 // [128][64] f16
  float* simw = (float*)(smem_raw + 16384);     // [64][128] f32
  h16* vws    = (h16*)(smem_raw + 49152);       // [128][64] f16
  h16* attw   = (h16*)smem_raw;                 // [64][128] f16 (reuses kws)
  int hh = blockIdx.x, w = blockIdx.y;
  int tid = threadIdx.x, lane = tid & 31, wave = tid >> 5;
  const h16* K = kh + (size_t)hh * NTOK * DH;
  const h16* V = vh + (size_t)hh * NTOK * DH;
  const h16* Q = qh + ((size_t)hh * NTOK + w * WIN) * DH;
  int kbase = (w - 1) * WIN;
  // stage Kw and Vw: 1024 x 16-byte chunks each, issued asynchronously
  for (int c = tid; c < 1024; c += 256) {
    int row = c >> 3, seg = c & 7;  // 8 chunks per 64-elem row
    int src = kbase + row;
    h16* ldk = kws + c * 8;
    h16* ldv = vws + c * 8;
    if (src >= 0) {
      async_g2l_b128(ldk, K + src * DH + seg * 8);
      async_g2l_b128(ldv, V + src * DH + seg * 8);
    } else {  // zero-pad window (w == 0)
      float4 z = {0.f, 0.f, 0.f, 0.f};
      *(float4*)ldk = z;
      *(float4*)ldv = z;
    }
  }
  wait_async0();
  __syncthreads();
  {  // sim = Q @ Kw^T : wave tile 16 x 64
    int m0 = (wave & 3) * 16, n0 = (wave >> 2) * 64;
    v8f acc[4] = {};
    for (int k0 = 0; k0 < DH; k0 += 32) {
      v16h a = fragA_rm(Q + m0 * DH + k0, DH, lane);
#pragma unroll
      for (int j = 0; j < 4; ++j) {
        v16h b = fragB_cm(kws + (n0 + 16 * j) * DH + k0, DH, lane);
        acc[j] = wmma32(a, b, acc[j]);
      }
    }
    int col0 = lane & 15, r0 = (lane & 16) ? 8 : 0;
#pragma unroll
    for (int j = 0; j < 4; ++j)
      for (int r = 0; r < 8; ++r)
        simw[(m0 + r0 + r) * 2 * WIN + n0 + 16 * j + col0] = acc[j][r] * SCALE;
  }
  __syncthreads();
  if (tid < WIN) {  // masked softmax per row
    int ti = tid;
    float mx = NEGF;
    for (int tj = 0; tj < 2 * WIN; ++tj) {
      int rel = ti - tj + WIN;
      bool ok = (rel >= 0) && (rel <= WIN) && ((w > 0) || (tj >= WIN));
      float v = ok ? simw[ti * 2 * WIN + tj] : NEGF;
      simw[ti * 2 * WIN + tj] = v;
      mx = fmaxf(mx, v);
    }
    float s = 0.f;
    for (int tj = 0; tj < 2 * WIN; ++tj) s += __expf(simw[ti * 2 * WIN + tj] - mx);
    float inv = 1.f / s;
    for (int tj = 0; tj < 2 * WIN; ++tj)
      attw[ti * 2 * WIN + tj] = (h16)(__expf(simw[ti * 2 * WIN + tj] - mx) * inv);
  }
  __syncthreads();
  {  // out = attn @ Vw : wave tile 16 x 32, K=128, B from LDS
    int m0 = (wave & 3) * 16, n0 = (wave >> 2) * 32;
    v8f acc[2] = {};
    for (int k0 = 0; k0 < 2 * WIN; k0 += 32) {
      v16h a = fragA_rm(attw + m0 * 2 * WIN + k0, 2 * WIN, lane);
#pragma unroll
      for (int j = 0; j < 2; ++j) {
        v16h b = fragB_rm(vws + k0 * DH + n0 + 16 * j, DH, lane);
        acc[j] = wmma32(a, b, acc[j]);
      }
    }
    float* dst = lout + ((size_t)hh * NTOK + w * WIN) * DH;
#pragma unroll
    for (int j = 0; j < 2; ++j)
      storeC_f32(dst + m0 * DH + n0 + 16 * j, DH, acc[j], lane);
  }
}

// ---------------------------------------------------------------------------
// K10: gates + merge three branches -> merged f16 (n, h*DH+d)
// ---------------------------------------------------------------------------
__global__ __launch_bounds__(256) void k_merge(const h16* xh, const h16* Wgt,
                                               const float* bg, const float* cout,
                                               const float* fout, const float* lout,
                                               h16* merged) {
  int n = blockIdx.x, tid = threadIdx.x;
  __shared__ float g[3 * H];
  if (tid < 3 * H) {
    const h16* xr = xh + (size_t)n * DIM;
    const h16* wr = Wgt + (size_t)tid * DIM;
    float acc = 0.f;
    for (int k = 0; k < DIM; ++k) acc += (float)xr[k] * (float)wr[k];
    acc += bg[tid];
    g[tid] = 1.f / (1.f + __expf(-acc));
  }
  __syncthreads();
  for (int c = tid; c < DIM; c += 256) {
    int hh = c / DH, d = c % DH;
    size_t off = ((size_t)hh * NTOK + n) * DH + d;
    float o = g[hh * 3 + 0] * cout[off] + g[hh * 3 + 1] * fout[off] +
              g[hh * 3 + 2] * lout[off];
    merged[(size_t)n * DIM + c] = (h16)o;
  }
}

// ---------------------------------------------------------------------------
// K11: output GEMM (1024 x 512 x 512) WMMA -> d_out (f32)
// ---------------------------------------------------------------------------
__global__ __launch_bounds__(256) void k_gemm_out(const h16* merged, const h16* Wt,
                                                  float* out) {
  int lane = threadIdx.x & 31, wave = threadIdx.x >> 5;
  int m0 = blockIdx.x * 128 + (wave & 3) * 32;
  int n0 = blockIdx.y * 128 + (wave >> 2) * 64;
  v8f acc[2][4] = {};
  for (int k0 = 0; k0 < DIM; k0 += 32) {
    __builtin_prefetch(merged + m0 * DIM + k0 + 64, 0, 1);
    v16h a[2], b[4];
#pragma unroll
    for (int i2 = 0; i2 < 2; ++i2)
      a[i2] = fragA_rm(merged + (m0 + 16 * i2) * DIM + k0, DIM, lane);
#pragma unroll
    for (int j = 0; j < 4; ++j)
      b[j] = fragB_cm(Wt + (n0 + 16 * j) * DIM + k0, DIM, lane);
#pragma unroll
    for (int i2 = 0; i2 < 2; ++i2)
#pragma unroll
      for (int j = 0; j < 4; ++j) acc[i2][j] = wmma32(a[i2], b[j], acc[i2][j]);
  }
#pragma unroll
  for (int i2 = 0; i2 < 2; ++i2)
#pragma unroll
    for (int j = 0; j < 4; ++j)
      storeC_f32(out + (size_t)(m0 + 16 * i2) * DIM + (n0 + 16 * j), DIM,
                 acc[i2][j], lane);
}

// ---------------------------------------------------------------------------
extern "C" void kernel_launch(void* const* d_in, const int* in_sizes, int n_in,
                              void* d_out, int out_size, void* d_ws, size_t ws_size,
                              hipStream_t stream) {
  (void)in_sizes; (void)n_in; (void)out_size; (void)ws_size;
  const float* inp   = (const float*)d_in[0];
  const float* normw = (const float*)d_in[1];
  const float* Wqkv  = (const float*)d_in[2];
  const float* kpos  = (const float*)d_in[3];
  const float* vpos  = (const float*)d_in[4];
  const float* kcw   = (const float*)d_in[5];
  const float* kcb   = (const float*)d_in[6];
  const float* vcw   = (const float*)d_in[7];
  const float* vcb   = (const float*)d_in[8];
  const float* memkv = (const float*)d_in[9];
  const float* Wg    = (const float*)d_in[10];
  const float* bg    = (const float*)d_in[11];
  const float* Wout  = (const float*)d_in[12];
  float* out = (float*)d_out;

  unsigned char* ws = (unsigned char*)d_ws;
  size_t off = 0;
  auto take = [&](size_t bytes) -> unsigned char* {
    unsigned char* p = ws + off;
    off = (off + bytes + 255) & ~(size_t)255;
    return p;
  };
  h16* xh     = (h16*)take((size_t)NTOK * DIM * 2);
  h16* Wqkvt  = (h16*)take((size_t)3 * DIM * DIM * 2);
  h16* Woutt  = (h16*)take((size_t)DIM * DIM * 2);
  h16* Wgt    = (h16*)take((size_t)3 * H * DIM * 2);
  h16* kcwt   = (h16*)take((size_t)H * DH * DH * BLK * 2);
  h16* vcwt   = (h16*)take((size_t)H * DH * DH * BLK * 2);
  h16* qh     = (h16*)take((size_t)H * NTOK * DH * 2);
  h16* kh     = (h16*)take((size_t)H * NTOK * DH * 2);
  h16* vh     = (h16*)take((size_t)H * NTOK * DH * 2);
  h16* ckh    = (h16*)take((size_t)H * NKCP * DH * 2);
  h16* cvh    = (h16*)take((size_t)H * NKCP * DH * 2);
  float* csim = (float*)take((size_t)H * NTOK * NKCP * 4);
  h16* catt   = (h16*)take((size_t)H * NTOK * NKCP * 2);
  int* selidx = (int*)take((size_t)H * NTOK * SEL * 4);
  unsigned* selflag = (unsigned*)take((size_t)H * NTOK * 4);
  float* cout = (float*)take((size_t)H * NTOK * DH * 4);
  float* fout = (float*)take((size_t)H * NTOK * DH * 4);
  float* lout = (float*)take((size_t)H * NTOK * DH * 4);
  h16* merged = (h16*)take((size_t)NTOK * DIM * 2);

  k_convert<<<dim3(512), dim3(256), 0, stream>>>(Wqkv, Wout, Wg, kcw, vcw, Wqkvt,
                                                 Woutt, Wgt, kcwt, vcwt);
  k_rmsnorm<<<dim3(NTOK), dim3(128), 0, stream>>>(inp, normw, xh);
  k_gemm_qkv<<<dim3(NTOK / 128, (3 * DIM) / 128), dim3(256), 0, stream>>>(
      xh, Wqkvt, qh, kh, vh);
  k_fill_mem<<<dim3(H), dim3(256), 0, stream>>>(memkv, ckh, cvh);
  k_compress<<<dim3(H), dim3(256), 0, stream>>>(kh, vh, kpos, vpos, kcb, vcb,
                                                kcwt, vcwt, ckh, cvh);
  k_csim<<<dim3(H, NTOK / 128), dim3(256), 0, stream>>>(qh, ckh, csim);
  k_csoftmax_topk<<<dim3((H * NTOK) / 256), dim3(256), 0, stream>>>(csim, catt,
                                                                    selidx, selflag);
  k_cout<<<dim3(H, NTOK / 128), dim3(256), 0, stream>>>(catt, cvh, cout);
  k_fine<<<dim3(H, NTOK / 8), dim3(256), 0, stream>>>(qh, kh, vh, selidx, selflag,
                                                      fout);
  k_local<<<dim3(H, NB), dim3(256), 0, stream>>>(qh, kh, vh, lout);
  k_merge<<<dim3(NTOK), dim3(256), 0, stream>>>(xh, Wgt, bg, cout, fout, lout,
                                                merged);
  k_gemm_out<<<dim3(NTOK / 128, DIM / 128), dim3(256), 0, stream>>>(merged, Woutt,
                                                                    out);
}
